// Generator_54339926229083
// MI455X (gfx1250) — matrix operand order repaired and checked by reading, not compiled, not run on hardware
//
#include <hip/hip_runtime.h>
#include <hip/hip_bf16.h>

typedef unsigned short u16;
typedef __attribute__((ext_vector_type(16))) __bf16 v16bf;
typedef __attribute__((ext_vector_type(8)))  float  v8f;

union V16 { uint4 q[2]; v16bf v; };

// ---------- bf16 helpers ----------
__device__ __forceinline__ u16 f2bf(float f) {
    unsigned u = __builtin_bit_cast(unsigned, f);
    u += 0x7FFFu + ((u >> 16) & 1u);
    return (u16)(u >> 16);
}
__device__ __forceinline__ float bf2f(u16 s) {
    unsigned u = ((unsigned)s) << 16;
    return __builtin_bit_cast(float, u);
}
// packed bf16 ReLU: one VOP3P instruction per dword
__device__ __forceinline__ unsigned relu2(unsigned x) {
    unsigned r;
    asm("v_pk_max_num_bf16 %0, %1, 0" : "=v"(r) : "v"(x));
    return r;
}
__device__ __forceinline__ uint4 relu4(uint4 v) {
    v.x = relu2(v.x); v.y = relu2(v.y); v.z = relu2(v.z); v.w = relu2(v.w);
    return v;
}

// =====================================================================
// Weight repack: f32 OIHW (64,64,3,3) -> bf16 WMMA-B fragments.
// Layer = 36864 shorts = 72KB. [tap 0..8][kc 0..1][nblk 0..3], 512 shorts each:
//   n = nblk*16 + (l&15);  K = (j<8 ? 8*(l>>4)+j : 16+8*(l>>4)+(j-8));  ci = kc*32 + K
// =====================================================================
__global__ void pack_conv_w(const float* __restrict__ src, u16* __restrict__ dst) {
    int layer = blockIdx.y;
    int e = blockIdx.x * 256 + threadIdx.x;           // grid.x = 144
    int frag = e >> 9, r = e & 511, lane = r >> 4, j = r & 15;
    int t  = frag >> 3, kc = (frag >> 2) & 1, nb = frag & 3;
    int n  = nb * 16 + (lane & 15);
    int K  = (j < 8) ? (8 * (lane >> 4) + j) : (16 + 8 * (lane >> 4) + (j - 8));
    int ci = kc * 32 + K;
    float v = src[(size_t)layer * 36864 + (size_t)n * 576 + ci * 9 + t];
    dst[(size_t)layer * 36864 + e] = f2bf(v);
}

// Deconv weights: ruw (Cin,Cout,4,4) -> per-class fragments.
// Layer = 65536 shorts. [class 0..3][tap 0..3][kc][nblk], 512 shorts each.
__global__ void pack_deconv_w(const float* __restrict__ src, u16* __restrict__ dst) {
    int layer = blockIdx.y;
    int e = blockIdx.x * 256 + threadIdx.x;           // grid.x = 256
    int frag = e >> 9, r = e & 511, lane = r >> 4, j = r & 15;
    int cls = frag >> 5, rem = frag & 31;
    int t  = rem >> 3, kc = (rem >> 2) & 1, nb = rem & 3;
    int n  = nb * 16 + (lane & 15);
    int K  = (j < 8) ? (8 * (lane >> 4) + j) : (16 + 8 * (lane >> 4) + (j - 8));
    int ci = kc * 32 + K;
    int cy = cls >> 1, cx = cls & 1, tyi = t >> 1, txi = t & 1;
    int ky = cy ? (tyi ? 2 : 0) : (tyi ? 3 : 1);
    int kx = cx ? (txi ? 2 : 0) : (txi ? 3 : 1);
    float v = src[(((size_t)layer * 64 + ci) * 64 + n) * 16 + ky * 4 + kx];
    dst[(size_t)layer * 65536 + e] = f2bf(v);
}

// =====================================================================
// Zero the 1-pixel halo border of a padded NHWC buffer [(H+2)x(W+2)x64].
// grid: ((P+255)/256, 8) where P = 2*(W+2) + 2*H perimeter pixels.
// =====================================================================
__global__ void zero_halo_k(u16* __restrict__ buf, int H, int W) {
    int b = blockIdx.y;
    int P = 2 * (W + 2) + 2 * H;
    int i = blockIdx.x * 256 + threadIdx.x;
    if (i >= P) return;
    int y, x;
    if (i < W + 2)               { y = 0;     x = i; }
    else if (i < 2 * (W + 2))    { y = H + 1; x = i - (W + 2); }
    else { int j = i - 2 * (W + 2); y = 1 + (j >> 1); x = (j & 1) ? (W + 1) : 0; }
    uint4* p = (uint4*)(buf + (((size_t)b * (H + 2) + y) * (W + 2) + x) * 64);
    uint4 z = make_uint4(0u, 0u, 0u, 0u);
    #pragma unroll
    for (int q = 0; q < 8; ++q) p[q] = z;
}

// =====================================================================
// 3x3 conv, Cin=Cout=64, implicit GEMM with bf16 WMMA.
// Padded NHWC [(H+2)x(W+2)x64] in/out; no bounds checks (zero halo).
// Tap addresses = one base pointer + compile-time immediate offsets.
// =====================================================================
template <int RELU, int HIN, int STRIDE>
__global__ __launch_bounds__(256) void conv3x3_k(
    const u16* __restrict__ in, const u16* __restrict__ wp,
    const float* __restrict__ bias, u16* __restrict__ out,
    const u16* __restrict__ resid)
{
    constexpr int H   = HIN;
    constexpr int Ho  = HIN / STRIDE;
    constexpr int Wo  = Ho;
    constexpr int WP  = HIN + 2;                  // padded input row stride
    constexpr int WOP = Wo + 2;                   // padded output row stride
    constexpr int SH  = (Wo >= 32) ? 5 : 4;
    constexpr int TW  = 1 << SH;
    constexpr int TPR = (Wo >= 32) ? (Wo / 32) : 1;

    extern __shared__ u16 shw[];                  // 36864 shorts = 72KB
    int b    = blockIdx.y;
    int tile = blockIdx.x;
    int lane = threadIdx.x & 31, wv = threadIdx.x >> 5;
    int mblk = wv & 1, nblk = wv >> 1;
    int row  = lane & 15, hi = lane >> 4;

    int ty0, tx0;
    if (Wo >= 32) { ty0 = tile / TPR; tx0 = (tile % TPR) << 5; }
    else          { ty0 = tile << 1;  tx0 = 0; }
    int m  = mblk * 16 + row;
    int py = ty0 + (m >> SH);
    int px = tx0 + (m & (TW - 1));

    // per-lane A base pointer at center tap (padded coords +1,+1)
    const u16* abase = in + (((size_t)b * (H + 2) + (py * STRIDE + 1)) * WP
                             + (px * STRIDE + 1)) * 64 + hi * 8;
    __builtin_prefetch(abase, 0, 0);

    // stage packed weights into LDS (4608 uint4 = 72KB)
    {
        const uint4* src = (const uint4*)wp;
        uint4* d4 = (uint4*)shw;
        for (int i = threadIdx.x; i < 4608; i += 256) d4[i] = src[i];
    }
    __syncthreads();

    auto loadTapA = [&](int t, V16& x0, V16& x1) {
        int dy = t / 3 - 1, dx = t % 3 - 1;
        const u16* ba = abase + (dy * WP + dx) * 64;   // imm offset after unroll
        x0.q[0] = *(const uint4*)(ba);
        x0.q[1] = *(const uint4*)(ba + 16);
        x1.q[0] = *(const uint4*)(ba + 32);
        x1.q[1] = *(const uint4*)(ba + 48);
        if (RELU) {
            x0.q[0] = relu4(x0.q[0]); x0.q[1] = relu4(x0.q[1]);
            x1.q[0] = relu4(x1.q[0]); x1.q[1] = relu4(x1.q[1]);
        }
    };
    auto loadTapB = [&](int t, V16& x0, V16& x1) {
        const u16* w0 = shw + (8 * t + nblk) * 512 + lane * 16;
        x0.q[0] = *(const uint4*)(w0);
        x0.q[1] = *(const uint4*)(w0 + 8);
        const u16* w1 = w0 + 4 * 512;
        x1.q[0] = *(const uint4*)(w1);
        x1.q[1] = *(const uint4*)(w1 + 8);
    };

    v8f acc = {0.f, 0.f, 0.f, 0.f, 0.f, 0.f, 0.f, 0.f};
    V16 aC0, aC1, bC0, bC1, aN0, aN1, bN0, bN1;
    loadTapA(0, aC0, aC1);
    loadTapB(0, bC0, bC1);
    #pragma unroll
    for (int t = 0; t < 9; ++t) {
        if (t + 1 < 9) {                          // issue next tap's loads first
            loadTapA(t + 1, aN0, aN1);
            loadTapB(t + 1, bN0, bN1);
        }
        acc = __builtin_amdgcn_wmma_f32_16x16x32_bf16(
            false, aC0.v, false, bC0.v, (short)0, acc, false, false);
        acc = __builtin_amdgcn_wmma_f32_16x16x32_bf16(
            false, aC1.v, false, bC1.v, (short)0, acc, false, false);
        aC0 = aN0; aC1 = aN1; bC0 = bN0; bC1 = bN1;
    }

    int n  = nblk * 16 + row;
    float bb = bias[n];
    #pragma unroll
    for (int r = 0; r < 8; ++r) {
        int mr  = mblk * 16 + hi * 8 + r;
        int py2 = ty0 + (mr >> SH);
        int px2 = tx0 + (mr & (TW - 1));
        size_t oidx = (((size_t)b * (Ho + 2) + (py2 + 1)) * WOP + (px2 + 1)) * 64 + n;
        float v = acc[r] + bb;
        if (resid) v += bf2f(resid[oidx]);
        out[oidx] = f2bf(v);
    }
}

// =====================================================================
// ConvTranspose2d(k=4,s=2,p=1) as 4 parity-class dense convs (K=256).
// blockIdx.y = class, blockIdx.z = batch. Padded in [(H+2)], out [(2H+2)].
// =====================================================================
template <int HIN>
__global__ __launch_bounds__(256) void deconv_k(
    const u16* __restrict__ in, const u16* __restrict__ wp,
    const float* __restrict__ bias, u16* __restrict__ out)
{
    constexpr int H   = HIN;
    constexpr int WP  = HIN + 2;
    constexpr int Ho  = 2 * HIN;
    constexpr int WOP = Ho + 2;
    constexpr int SH  = (HIN >= 32) ? 5 : 4;
    constexpr int TW  = 1 << SH;
    constexpr int TPR = (HIN >= 32) ? (HIN / 32) : 1;

    extern __shared__ u16 shw[];                  // 16384 shorts = 32KB
    int cls = blockIdx.y, b = blockIdx.z;
    {
        const uint4* src = (const uint4*)(wp + (size_t)cls * 16384);
        uint4* d4 = (uint4*)shw;
        for (int i = threadIdx.x; i < 2048; i += 256) d4[i] = src[i];
    }

    int lane = threadIdx.x & 31, wv = threadIdx.x >> 5;
    int mblk = wv & 1, nblk = wv >> 1;
    int row  = lane & 15, hi = lane >> 4;
    int tile = blockIdx.x;
    int ty0, tx0;
    if (HIN >= 32) { ty0 = tile / TPR; tx0 = (tile % TPR) << 5; }
    else           { ty0 = tile << 1;  tx0 = 0; }
    int m  = mblk * 16 + row;
    int py = ty0 + (m >> SH);
    int px = tx0 + (m & (TW - 1));
    int cy = cls >> 1, cx = cls & 1;

    const u16* abase = in + (((size_t)b * (H + 2) + (py + 1)) * WP + (px + 1)) * 64 + hi * 8;
    const u16* tap[4];
    #pragma unroll
    for (int t = 0; t < 4; ++t) {
        int tyi = t >> 1, txi = t & 1;
        int dy = cy ? (1 - tyi) : -tyi;
        int dx = cx ? (1 - txi) : -txi;
        tap[t] = abase + (dy * WP + dx) * 64;
    }
    __syncthreads();

    auto loadTapA = [&](int t, V16& x0, V16& x1) {
        const u16* ba = tap[t];
        x0.q[0] = *(const uint4*)(ba);
        x0.q[1] = *(const uint4*)(ba + 16);
        x1.q[0] = *(const uint4*)(ba + 32);
        x1.q[1] = *(const uint4*)(ba + 48);
    };
    auto loadTapB = [&](int t, V16& x0, V16& x1) {
        const u16* w0 = shw + (8 * t + nblk) * 512 + lane * 16;
        x0.q[0] = *(const uint4*)(w0);
        x0.q[1] = *(const uint4*)(w0 + 8);
        const u16* w1 = w0 + 4 * 512;
        x1.q[0] = *(const uint4*)(w1);
        x1.q[1] = *(const uint4*)(w1 + 8);
    };

    v8f acc = {0.f, 0.f, 0.f, 0.f, 0.f, 0.f, 0.f, 0.f};
    V16 aC0, aC1, bC0, bC1, aN0, aN1, bN0, bN1;
    loadTapA(0, aC0, aC1);
    loadTapB(0, bC0, bC1);
    #pragma unroll
    for (int t = 0; t < 4; ++t) {
        if (t + 1 < 4) {
            loadTapA(t + 1, aN0, aN1);
            loadTapB(t + 1, bN0, bN1);
        }
        acc = __builtin_amdgcn_wmma_f32_16x16x32_bf16(
            false, aC0.v, false, bC0.v, (short)0, acc, false, false);
        acc = __builtin_amdgcn_wmma_f32_16x16x32_bf16(
            false, aC1.v, false, bC1.v, (short)0, acc, false, false);
        aC0 = aN0; aC1 = aN1; bC0 = bN0; bC1 = bN1;
    }

    int n = nblk * 16 + row;
    float bb = bias[n];
    #pragma unroll
    for (int r = 0; r < 8; ++r) {
        int mr  = mblk * 16 + hi * 8 + r;
        int py2 = ty0 + (mr >> SH);
        int px2 = tx0 + (mr & (TW - 1));
        int oy = py2 * 2 + cy, ox = px2 * 2 + cx;
        out[(((size_t)b * (Ho + 2) + (oy + 1)) * WOP + (ox + 1)) * 64 + n] = f2bf(acc[r] + bb);
    }
}

// =====================================================================
// Fused corrcoef skip on padded buffers: dst = a + b*(cor>0?cor:1).
// grid = (64, B); maps are (2^logW) square.
// =====================================================================
__global__ void corr_add_k(const u16* __restrict__ a, const u16* __restrict__ bm,
                           u16* __restrict__ dst, int logW)
{
    __shared__ float red[5 * 256];
    __shared__ float scale_sh;
    int W = 1 << logW, HW = W * W, WP = W + 2;
    int c = blockIdx.x, b = blockIdx.y;
    auto idx = [&](int i) {
        int y = i >> logW, x = i & (W - 1);
        return (((size_t)b * (W + 2) + (y + 1)) * WP + (x + 1)) * 64 + c;
    };
    float sa = 0.f, sb = 0.f, saa = 0.f, sbb = 0.f, sab = 0.f;
    for (int i = threadIdx.x; i < HW; i += 256) {
        size_t o = idx(i);
        float av = bf2f(a[o]);
        float bv = bf2f(bm[o]);
        sa += av; sb += bv; saa += av * av; sbb += bv * bv; sab += av * bv;
    }
    red[threadIdx.x]        = sa;
    red[256 + threadIdx.x]  = sb;
    red[512 + threadIdx.x]  = saa;
    red[768 + threadIdx.x]  = sbb;
    red[1024 + threadIdx.x] = sab;
    __syncthreads();
    for (int s = 128; s > 0; s >>= 1) {
        if (threadIdx.x < (unsigned)s)
            for (int k = 0; k < 5; ++k)
                red[k * 256 + threadIdx.x] += red[k * 256 + threadIdx.x + s];
        __syncthreads();
    }
    if (threadIdx.x == 0) {
        float N  = (float)HW;
        float Sa = red[0], Sb = red[256], Saa = red[512], Sbb = red[768], Sab = red[1024];
        float cov = Sab - Sa * Sb / N;
        float va  = Saa - Sa * Sa / N;
        float vb  = Sbb - Sb * Sb / N;
        float cor = cov * rsqrtf(va * vb);
        scale_sh = (cor > 0.f) ? cor : 1.f;
    }
    __syncthreads();
    float sc = scale_sh;
    for (int i = threadIdx.x; i < HW; i += 256) {
        size_t o = idx(i);
        dst[o] = f2bf(bf2f(a[o]) + bf2f(bm[o]) * sc);
    }
}

// ---------- edge convs ----------
// x f32 NCHW (8,3,64,64) -> bf16 padded NHWC (8,66,66,64)
__global__ void conv_in_k(const float* __restrict__ x, const float* __restrict__ w,
                          const float* __restrict__ bias, u16* __restrict__ out)
{
    __shared__ float sw[64 * 27];
    for (int i = threadIdx.x; i < 64 * 27; i += 256) sw[i] = w[i];
    __syncthreads();
    int b = blockIdx.y;
    int p = blockIdx.x * 256 + threadIdx.x;           // 0..4095
    int py = p >> 6, px = p & 63;
    float inv[27];
    int k = 0;
    for (int ci = 0; ci < 3; ++ci)
        for (int dy = -1; dy <= 1; ++dy)
            for (int dx = -1; dx <= 1; ++dx) {
                int iy = py + dy, ix = px + dx;
                inv[k++] = (iy >= 0 && iy < 64 && ix >= 0 && ix < 64)
                           ? x[(((size_t)b * 3 + ci) * 64 + iy) * 64 + ix] : 0.f;
            }
    u16* op = out + (((size_t)b * 66 + (py + 1)) * 66 + (px + 1)) * 64;
    for (int co = 0; co < 64; ++co) {
        float s = bias[co];
        const float* wc = sw + co * 27;
        #pragma unroll
        for (int q = 0; q < 27; ++q) s += inv[q] * wc[q];
        op[co] = f2bf(s);
    }
}

// bf16 padded NHWC (8,258,258,64) -> f32 NCHW (8,3,256,256); halo = zeros
__global__ void conv_out_k(const u16* __restrict__ in, const float* __restrict__ w,
                           const float* __restrict__ bias, float* __restrict__ out)
{
    __shared__ float sw[3 * 576];
    for (int i = threadIdx.x; i < 3 * 576; i += 256) sw[i] = w[i];
    __syncthreads();
    int b = blockIdx.y;
    int p = blockIdx.x * 256 + threadIdx.x;           // 0..65535
    int py = p >> 8, px = p & 255;
    float s0 = bias[0], s1 = bias[1], s2 = bias[2];
    #pragma unroll
    for (int dy = -1; dy <= 1; ++dy) {
        #pragma unroll
        for (int dx = -1; dx <= 1; ++dx) {
            const u16* ip = in + (((size_t)b * 258 + (py + 1 + dy)) * 258
                                  + (px + 1 + dx)) * 64;
            int t = (dy + 1) * 3 + (dx + 1);
            for (int ci = 0; ci < 64; ++ci) {
                float f = bf2f(ip[ci]);
                s0 += f * sw[0 * 576 + ci * 9 + t];
                s1 += f * sw[1 * 576 + ci * 9 + t];
                s2 += f * sw[2 * 576 + ci * 9 + t];
            }
        }
    }
    size_t o = (size_t)b * 3 * 65536 + p;
    out[o] = s0; out[o + 65536] = s1; out[o + 2 * 65536] = s2;
}

// ---------- host-side templated launchers ----------
template <int RELU, int HIN, int STRIDE>
static void launch_conv(const u16* in, const u16* wp, const float* bias, u16* out,
                        const u16* resid, hipStream_t s)
{
    constexpr int Ho = HIN / STRIDE;
    constexpr int tiles = Ho * Ho / 32;
    conv3x3_k<RELU, HIN, STRIDE><<<dim3(tiles, 8), 256, 73728, s>>>(in, wp, bias, out, resid);
}
template <int HIN>
static void launch_deconv(const u16* in, const u16* wp, const float* bias, u16* out,
                          hipStream_t s)
{
    constexpr int tiles = HIN * HIN / 32;
    deconv_k<HIN><<<dim3(tiles, 4, 8), 256, 32768, s>>>(in, wp, bias, out);
}
template <int HIN>
static void run_group(u16* X, u16* P, u16* Q, const u16* cp, int L0,
                      const float* bias0, int nunits, hipStream_t s)
{
    for (int u = 0; u < nunits; ++u) {
        int L = L0 + u * 4;
        const float* bb = bias0 + (size_t)u * 4 * 64;
        launch_conv<1, HIN, 1>(X, cp + (size_t)(L + 0) * 36864, bb + 0,   P, nullptr, s);
        launch_conv<1, HIN, 1>(P, cp + (size_t)(L + 1) * 36864, bb + 64,  Q, nullptr, s);
        launch_conv<1, HIN, 1>(Q, cp + (size_t)(L + 2) * 36864, bb + 128, P, nullptr, s);
        launch_conv<1, HIN, 1>(P, cp + (size_t)(L + 3) * 36864, bb + 192, X, X, s);
    }
}
static void zero_halo(u16* buf, int H, hipStream_t s)
{
    int P = 2 * (H + 2) + 2 * H;
    zero_halo_k<<<dim3((P + 255) / 256, 8), 256, 0, s>>>(buf, H, H);
}

// =====================================================================
extern "C" void kernel_launch(void* const* d_in, const int* in_sizes, int n_in,
                              void* d_out, int out_size, void* d_ws, size_t ws_size,
                              hipStream_t stream)
{
    (void)in_sizes; (void)n_in; (void)out_size; (void)ws_size;

    const float* x       = (const float*)d_in[0];
    const float* w_proc  = (const float*)d_in[1];
    const float* b_proc  = (const float*)d_in[2];
    const float* gw      = (const float*)d_in[3];
    const float* gb      = (const float*)d_in[4];
    const float* g6w     = (const float*)d_in[5];
    const float* g6b     = (const float*)d_in[6];
    const float* rdw     = (const float*)d_in[7];
    const float* rdb     = (const float*)d_in[8];
    const float* ruw     = (const float*)d_in[9];
    const float* rub     = (const float*)d_in[10];
    const float* w_final = (const float*)d_in[11];
    const float* b_final = (const float*)d_in[12];

    // allow >64KB dynamic LDS on every instantiation
    (void)hipFuncSetAttribute(reinterpret_cast<const void*>(conv3x3_k<1, 16, 1>),
                              hipFuncAttributeMaxDynamicSharedMemorySize, 73728);
    (void)hipFuncSetAttribute(reinterpret_cast<const void*>(conv3x3_k<1, 32, 1>),
                              hipFuncAttributeMaxDynamicSharedMemorySize, 73728);
    (void)hipFuncSetAttribute(reinterpret_cast<const void*>(conv3x3_k<1, 64, 1>),
                              hipFuncAttributeMaxDynamicSharedMemorySize, 73728);
    (void)hipFuncSetAttribute(reinterpret_cast<const void*>(conv3x3_k<1, 128, 1>),
                              hipFuncAttributeMaxDynamicSharedMemorySize, 73728);
    (void)hipFuncSetAttribute(reinterpret_cast<const void*>(conv3x3_k<1, 256, 1>),
                              hipFuncAttributeMaxDynamicSharedMemorySize, 73728);
    (void)hipFuncSetAttribute(reinterpret_cast<const void*>(conv3x3_k<0, 64, 2>),
                              hipFuncAttributeMaxDynamicSharedMemorySize, 73728);
    (void)hipFuncSetAttribute(reinterpret_cast<const void*>(conv3x3_k<0, 32, 2>),
                              hipFuncAttributeMaxDynamicSharedMemorySize, 73728);
    (void)hipFuncSetAttribute(reinterpret_cast<const void*>(deconv_k<16>),
                              hipFuncAttributeMaxDynamicSharedMemorySize, 32768);
    (void)hipFuncSetAttribute(reinterpret_cast<const void*>(deconv_k<32>),
                              hipFuncAttributeMaxDynamicSharedMemorySize, 32768);
    (void)hipFuncSetAttribute(reinterpret_cast<const void*>(deconv_k<64>),
                              hipFuncAttributeMaxDynamicSharedMemorySize, 32768);
    (void)hipFuncSetAttribute(reinterpret_cast<const void*>(deconv_k<128>),
                              hipFuncAttributeMaxDynamicSharedMemorySize, 32768);

    const size_t MB = 1ull << 20;
    u16*  cp    = (u16*)d_ws;                          // 54 * 36864 shorts (~3.8MB)
    u16*  dp    = (u16*)((char*)d_ws + 4 * MB);        // 4 * 65536 shorts (0.5MB)
    char* arena = (char*)d_ws + 8 * MB;                // 203MB activation arena

    // padded sizes: 258^2*64*2*8 = 65.0MiB ; 130^2 = 16.5MiB ; 66^2 = 4.26MiB ;
    //               34^2 = 1.13MiB ; 18^2 = 0.32MiB
    u16* B256_0 = (u16*)(arena);
    u16* B256_1 = (u16*)(arena + 69 * MB);
    u16* B256_2 = (u16*)(arena + 138 * MB);
    // 128-scale trio aliases B256_1 span (dead before group6's 1st conv)
    u16* F128_0 = (u16*)(arena + 69 * MB);
    u16* F128_1 = (u16*)(arena + 86 * MB);
    u16* F128_2 = (u16*)(arena + 103 * MB);
    // small scales alias B256_0 span (dead before ru3 writes it)
    u16* F64_0 = (u16*)(arena + 0 * MB);
    u16* F64_1 = (u16*)(arena + 5 * MB);
    u16* F64_2 = (u16*)(arena + 10 * MB);
    u16* F32_0 = (u16*)(arena + 15 * MB);
    u16* F32_1 = (u16*)(arena + 17 * MB);
    u16* F32_2 = (u16*)(arena + 19 * MB);
    u16* F16_0 = (u16*)(arena + 21 * MB);
    u16* F16_1 = (u16*)(arena + 22 * MB);
    u16* F16_2 = (u16*)(arena + 23 * MB);

    // ---- repack weights (every call; deterministic, graph-safe) ----
    pack_conv_w<<<dim3(144, 48), 256, 0, stream>>>(gw,  cp);                    // 0..47
    pack_conv_w<<<dim3(144, 4),  256, 0, stream>>>(g6w, cp + 48ull * 36864);    // 48..51
    pack_conv_w<<<dim3(144, 2),  256, 0, stream>>>(rdw, cp + 52ull * 36864);    // 52..53
    pack_deconv_w<<<dim3(256, 4), 256, 0, stream>>>(ruw, dp);

    // ---- zero halos (initial set; aliased 256-halos deferred) ----
    zero_halo(F64_0, 64, stream);  zero_halo(F64_1, 64, stream);  zero_halo(F64_2, 64, stream);
    zero_halo(F32_0, 32, stream);  zero_halo(F32_1, 32, stream);  zero_halo(F32_2, 32, stream);
    zero_halo(F16_0, 16, stream);  zero_halo(F16_1, 16, stream);  zero_halo(F16_2, 16, stream);
    zero_halo(F128_0, 128, stream); zero_halo(F128_1, 128, stream); zero_halo(F128_2, 128, stream);
    zero_halo(B256_2, 256, stream);

    // ---- forward pass ----
    conv_in_k<<<dim3(16, 8), 256, 0, stream>>>(x, w_proc, b_proc, F64_0);

    run_group<64>(F64_0, F64_1, F64_2, cp, 0, gb, 2, stream);                   // out0 = F64_0
    launch_conv<0, 64, 2>(F64_0, cp + 52ull * 36864, rdb, F32_0, nullptr, stream);
    run_group<32>(F32_0, F32_1, F32_2, cp, 8, gb + 512, 2, stream);             // out1 = F32_0
    launch_conv<0, 32, 2>(F32_0, cp + 53ull * 36864, rdb + 64, F16_0, nullptr, stream);
    run_group<16>(F16_0, F16_1, F16_2, cp, 16, gb + 1024, 2, stream);           // out2 = F16_0

    launch_deconv<16>(F16_0, dp, rub, F32_1, stream);                           // ru0 -> 32
    corr_add_k<<<dim3(64, 8), 256, 0, stream>>>(F32_1, F32_0, F32_2, 5);
    run_group<32>(F32_2, F32_0, F32_1, cp, 24, gb + 1536, 2, stream);           // out3 = F32_2

    launch_deconv<32>(F32_2, dp + 1ull * 65536, rub + 64, F64_1, stream);       // ru1 -> 64
    corr_add_k<<<dim3(64, 8), 256, 0, stream>>>(F64_1, F64_0, F64_2, 6);
    run_group<64>(F64_2, F64_0, F64_1, cp, 32, gb + 2048, 2, stream);           // out4 = F64_2

    launch_deconv<64>(F64_2, dp + 2ull * 65536, rub + 128, F128_0, stream);     // ru2 -> 128
    zero_halo(B256_0, 256, stream);            // small-scale region now dead
    run_group<128>(F128_0, F128_1, F128_2, cp, 40, gb + 2560, 2, stream);       // out5 = F128_0

    launch_deconv<128>(F128_0, dp + 3ull * 65536, rub + 192, B256_0, stream);   // ru3 -> 256
    zero_halo(B256_1, 256, stream);            // F128 region now dead
    run_group<256>(B256_0, B256_1, B256_2, cp, 48, g6b, 1, stream);             // group6

    conv_out_k<<<dim3(256, 8), 256, 0, stream>>>(B256_0, w_final, b_final, (float*)d_out);
}
